// GaussianAttention_11965778886676
// MI455X (gfx1250) — matrix operand order repaired and checked by reading, not compiled, not run on hardware
//
#include <hip/hip_runtime.h>

#define S_LEN 2048
#define D_DIM 64
#define BM 64      // query rows per workgroup (16 per wave, 4 waves)
#define BN 64      // key chunk
#define KSTR 72    // LDS row stride (elements): 144B rows -> conflict-free b128 frags

#define LOG2E  1.4426950408889634f
#define LOG2E2 2.0813689810056077f   // log2(e)^2

typedef __attribute__((ext_vector_type(8)))  float    v8f;
typedef __attribute__((ext_vector_type(8)))  __bf16   v8bf;
typedef __attribute__((ext_vector_type(16))) __bf16   v16bf;
typedef __attribute__((ext_vector_type(8)))  _Float16 v8h;
typedef __attribute__((ext_vector_type(16))) _Float16 v16h;

static __device__ inline v8f zero8() {
  v8f z;
#pragma unroll
  for (int i = 0; i < 8; ++i) z[i] = 0.0f;
  return z;
}

// A-fragment (16x32, 16-bit) from row-major LDS tile, row stride KSTR.
// lanes 0-15: row=lane,   K = kc+{0..7, 16..23}
// lanes 16-31: row=lane-16, K = kc+{8..15, 24..31}
static __device__ inline v16bf load_a_bf(const __bf16* rowbase, int kc, int lane) {
  const __bf16* p = rowbase + (lane & 15) * KSTR + kc + ((lane >> 4) << 3);
  union { v16bf v; v8bf h[2]; } u;
  u.h[0] = *(const v8bf*)(p);
  u.h[1] = *(const v8bf*)(p + 16);
  return u.v;
}
static __device__ inline v16h load_a_h(const _Float16* rowbase, int kc, int lane) {
  const _Float16* p = rowbase + (lane & 15) * KSTR + kc + ((lane >> 4) << 3);
  union { v16h v; v8h h[2]; } u;
  u.h[0] = *(const v8h*)(p);
  u.h[1] = *(const v8h*)(p + 16);
  return u.v;
}
// B-fragment (32x16, 16-bit): lane holds column n = nb+(lane&15),
// K = kc + (lane>=16 ? 8 : 0) + {0..7, 16..23}, from a tile whose row n
// stores the K dimension contiguously (K-tile row-major / V transposed).
static __device__ inline v16bf load_b_bf(const __bf16* tile, int nb, int kc, int lane) {
  const __bf16* p = tile + (nb + (lane & 15)) * KSTR + kc + ((lane >> 4) << 3);
  union { v16bf v; v8bf h[2]; } u;
  u.h[0] = *(const v8bf*)(p);
  u.h[1] = *(const v8bf*)(p + 16);
  return u.v;
}
static __device__ inline v16h load_b_h(const _Float16* tile, int nb, int kc, int lane) {
  const _Float16* p = tile + (nb + (lane & 15)) * KSTR + kc + ((lane >> 4) << 3);
  union { v16h v; v8h h[2]; } u;
  u.h[0] = *(const v8h*)(p);
  u.h[1] = *(const v8h*)(p + 16);
  return u.v;
}

__global__ __launch_bounds__(128)
void gaussian_attn_wmma(const float* __restrict__ Q, const float* __restrict__ K,
                        const float* __restrict__ V, float* __restrict__ O) {
  __shared__ __bf16   s_qhi[BM * KSTR];
  __shared__ __bf16   s_qlo[BM * KSTR];
  __shared__ __bf16   s_khi[BN * KSTR];
  __shared__ __bf16   s_klo[BN * KSTR];
  __shared__ _Float16 s_vt [D_DIM * KSTR];   // V transposed: [col][key]
  __shared__ _Float16 s_p  [BM * KSTR];      // per-wave 16x64 P scratch
  __shared__ float    s_q2 [BM];             // log2(e)^2 * ||q||^2
  __shared__ float    s_k2 [BN];             // log2(e)^2 * ||k||^2

  const int tid  = threadIdx.x;
  const int lane = tid & 31;
  const int wave = tid >> 5;
  const int b    = blockIdx.x >> 5;   // 32 row-tiles per batch
  const int mt   = blockIdx.x & 31;

  const float* Qg = Q + ((size_t)b * S_LEN + (size_t)mt * BM) * D_DIM;
  const float* Kg = K + (size_t)b * S_LEN * D_DIM;
  const float* Vg = V + (size_t)b * S_LEN * D_DIM;
  float*       Og = O + ((size_t)b * S_LEN + (size_t)mt * BM) * D_DIM;

  // ---- stage Q as bf16 hi/lo split; exact fp32 q2 (pre-scaled) ----
  {
    const int row = tid >> 1, cb = (tid & 1) * 32;
    const float4* src = (const float4*)(Qg + row * D_DIM + cb);
    __bf16* hi = s_qhi + row * KSTR + cb;
    __bf16* lo = s_qlo + row * KSTR + cb;
#pragma unroll
    for (int j = 0; j < 8; ++j) {
      float4 x = src[j];
      float f[4] = {x.x, x.y, x.z, x.w};
#pragma unroll
      for (int e = 0; e < 4; ++e) {
        __bf16 h = (__bf16)f[e];
        hi[j * 4 + e] = h;
        lo[j * 4 + e] = (__bf16)(f[e] - (float)h);
      }
    }
  }
  if (tid < BM) {
    const float* r = Qg + tid * D_DIM;
    float s = 0.0f;
#pragma unroll 8
    for (int j = 0; j < D_DIM; ++j) s += r[j] * r[j];
    s_q2[tid] = s * LOG2E2;
  }
  __syncthreads();

  v8f q2v;
#pragma unroll
  for (int i = 0; i < 8; ++i) q2v[i] = s_q2[wave * 16 + ((lane >> 4) << 3) + i];

  v8f rowsum = zero8();
  v8f oacc[4];
#pragma unroll
  for (int c = 0; c < 4; ++c) oacc[c] = zero8();

  const __bf16* qhi_w = s_qhi + wave * 16 * KSTR;
  const __bf16* qlo_w = s_qlo + wave * 16 * KSTR;
  _Float16*     pw    = s_p   + wave * 16 * KSTR;

  for (int kb = 0; kb < S_LEN / BN; ++kb) {
    const float* Kc = Kg + (size_t)kb * BN * D_DIM;
    const float* Vc = Vg + (size_t)kb * BN * D_DIM;

    // ---- stage K chunk (bf16 hi/lo) and V chunk (f16, transposed) ----
    {
      const int row = tid >> 1, cb = (tid & 1) * 32;
      const float4* ks = (const float4*)(Kc + row * D_DIM + cb);
      const float4* vs = (const float4*)(Vc + row * D_DIM + cb);
      __bf16* hi = s_khi + row * KSTR + cb;
      __bf16* lo = s_klo + row * KSTR + cb;
#pragma unroll
      for (int j = 0; j < 8; ++j) {
        float4 x = ks[j];
        float f[4] = {x.x, x.y, x.z, x.w};
#pragma unroll
        for (int e = 0; e < 4; ++e) {
          __bf16 h = (__bf16)f[e];
          hi[j * 4 + e] = h;
          lo[j * 4 + e] = (__bf16)(f[e] - (float)h);
        }
        float4 y = vs[j];
        float g[4] = {y.x, y.y, y.z, y.w};
#pragma unroll
        for (int e = 0; e < 4; ++e)
          s_vt[(cb + j * 4 + e) * KSTR + row] = (_Float16)g[e];
      }
      // speculative prefetch of next chunk (silently dropped if OOB)
      __builtin_prefetch(Kc + BN * D_DIM + row * D_DIM + cb, 0, 0);
      __builtin_prefetch(Vc + BN * D_DIM + row * D_DIM + cb, 0, 0);
    }
    if (tid < BN) {
      const float* r = Kc + tid * D_DIM;
      float s = 0.0f;
#pragma unroll 8
      for (int j = 0; j < D_DIM; ++j) s += r[j] * r[j];
      s_k2[tid] = s * LOG2E2;
    }
    __syncthreads();

    // ---- scores via bf16x3 WMMA ----
    // d2' = log2e^2 * ||q-k||^2  ->  w = exp2(-sqrt(d2')) = exp(-dist)
    // p  = exp2(log2e * w) = exp(w); w in (0,1] so softmax needs no max.
#pragma unroll
    for (int t4 = 0; t4 < 4; ++t4) {
      v8f acc = zero8();
#pragma unroll
      for (int kc = 0; kc < D_DIM; kc += 32) {
        v16bf ahi = load_a_bf(qhi_w, kc, lane);
        v16bf alo = load_a_bf(qlo_w, kc, lane);
        v16bf bhi = load_b_bf(s_khi, t4 * 16, kc, lane);
        v16bf blo = load_b_bf(s_klo, t4 * 16, kc, lane);
        acc = __builtin_amdgcn_wmma_f32_16x16x32_bf16(false, ahi, false, bhi, (short)0, acc, false, false);
        acc = __builtin_amdgcn_wmma_f32_16x16x32_bf16(false, ahi, false, blo, (short)0, acc, false, false);
        acc = __builtin_amdgcn_wmma_f32_16x16x32_bf16(false, alo, false, bhi, (short)0, acc, false, false);
      }
      const float k2n = s_k2[t4 * 16 + (lane & 15)];
      // Stage-split vector phases: 8 independent ops per stage so the
      // pipelined TRANS unit never stalls on its own latency.
      v8f dv, wv, prv;
#pragma unroll
      for (int i = 0; i < 8; ++i)
        dv[i] = fmaxf(fmaf(acc[i], -2.0f * LOG2E2, q2v[i] + k2n), 0.0f);
#pragma unroll
      for (int i = 0; i < 8; ++i) dv[i] = __builtin_amdgcn_sqrtf(dv[i]);
#pragma unroll
      for (int i = 0; i < 8; ++i) wv[i] = __builtin_amdgcn_exp2f(-dv[i]);   // exp(-dist)
#pragma unroll
      for (int i = 0; i < 8; ++i) prv[i] = __builtin_amdgcn_exp2f(LOG2E * wv[i]); // exp(w)
#pragma unroll
      for (int i = 0; i < 8; ++i) rowsum[i] += prv[i];
#pragma unroll
      for (int i = 0; i < 8; ++i)
        pw[(((lane >> 4) << 3) + i) * KSTR + t4 * 16 + (lane & 15)] = (_Float16)prv[i];
    }
    // Wave-private LDS RAW (P scratch): same-wave LDS ops execute in order,
    // so no counter wait is required -- just stop the compiler reordering.
    asm volatile("" ::: "memory");

    // ---- O += P * V via f16 WMMA ----
#pragma unroll
    for (int pc = 0; pc < BN; pc += 32) {
      v16h ap = load_a_h(pw, pc, lane);
#pragma unroll
      for (int c = 0; c < 4; ++c) {
        v16h bv = load_b_h(s_vt, c * 16, pc, lane);
        oacc[c] = __builtin_amdgcn_wmma_f32_16x16x32_f16(false, ap, false, bv, (short)0, oacc[c], false, false);
      }
    }
    __syncthreads();
  }

  // ---- reduce denominators across each 16-lane half, normalize, store ----
#pragma unroll
  for (int i = 0; i < 8; ++i) {
    float s = rowsum[i];
    s += __shfl_xor(s, 1, 32);
    s += __shfl_xor(s, 2, 32);
    s += __shfl_xor(s, 4, 32);
    s += __shfl_xor(s, 8, 32);
    const float inv = 1.0f / s;
    const int row = wave * 16 + ((lane >> 4) << 3) + i;
#pragma unroll
    for (int c = 0; c < 4; ++c)
      Og[row * D_DIM + c * 16 + (lane & 15)] = oacc[c][i] * inv;
  }
}

extern "C" void kernel_launch(void* const* d_in, const int* in_sizes, int n_in,
                              void* d_out, int out_size, void* d_ws, size_t ws_size,
                              hipStream_t stream) {
  (void)in_sizes; (void)n_in; (void)out_size; (void)d_ws; (void)ws_size;
  const float* q = (const float*)d_in[0];
  const float* k = (const float*)d_in[1];
  const float* v = (const float*)d_in[2];
  float*       o = (float*)d_out;
  dim3 grid(8 * (S_LEN / BM));   // 256 workgroups
  dim3 block(128);               // 4 wave32
  gaussian_attn_wmma<<<grid, block, 0, stream>>>(q, k, v, o);
}
